// TemporalContrastiveSAE_90984587198475
// MI455X (gfx1250) — matrix operand order repaired and checked by reading, not compiled, not run on hardware
//
#include <hip/hip_runtime.h>
#include <hip/hip_bf16.h>
#include <math.h>
#include <stdint.h>

// ---------------------------------------------------------------------------
// TemporalContrastiveSAE forward on MI455X (gfx1250, wave32, WMMA bf16,
// async global->LDS tile DMA with ASYNCcnt double buffering)
// ---------------------------------------------------------------------------

typedef __attribute__((ext_vector_type(16))) __bf16 v16bf;
typedef __attribute__((ext_vector_type(8)))  __bf16 bf16x8;
typedef __attribute__((ext_vector_type(8)))  float  v8f;

#define DIN   2048
#define DSAE  16384
#define NROW  4096   // B * 2 tokens
#define BPAIR 2048   // B
#define HDIM  8192
#define KTOP  64
#define EPSN  1e-8f

// ---------------- CDNA5 async global->LDS helpers (inline asm) -------------
__device__ __forceinline__ void async_b128_to_lds(void* lds_dst,
                                                  const void* gsrc) {
  uint32_t l = (uint32_t)(uintptr_t)lds_dst;   // low 32b of generic = LDS off
  uint64_t g = (uint64_t)(uintptr_t)gsrc;
  asm volatile("global_load_async_to_lds_b128 %0, %1, off"
               :: "v"(l), "v"(g) : "memory");
}
__device__ __forceinline__ void wait_async0() {
  asm volatile("s_wait_asynccnt 0x0" ::: "memory");
}

// ----------------------------- utility kernels -----------------------------
__global__ void sae_cast_f32_bf16(const float* __restrict__ in,
                                  __bf16* __restrict__ out, size_t n) {
  size_t i = (size_t)blockIdx.x * blockDim.x + threadIdx.x;
  size_t stride = (size_t)gridDim.x * blockDim.x;
  for (; i < n; i += stride) out[i] = (__bf16)in[i];
}

__global__ void sae_zero_f32(float* __restrict__ p, size_t n) {
  size_t i = (size_t)blockIdx.x * blockDim.x + threadIdx.x;
  size_t stride = (size_t)gridDim.x * blockDim.x;
  for (; i < n; i += stride) p[i] = 0.0f;
}

// cast f32 [Kd][Nd] -> bf16 [Nd][Kd]  (tiled 32x32 transpose through LDS)
__global__ __launch_bounds__(256)
void sae_cast_transpose(const float* __restrict__ in, __bf16* __restrict__ out,
                        int Kd, int Nd) {
  __shared__ __bf16 t[32][33];
  const int bx = blockIdx.x * 32;            // N tile
  const int by = blockIdx.y * 32;            // K tile
  const int tx = threadIdx.x & 31;
  const int ty = threadIdx.x >> 5;           // 8 row-groups
#pragma unroll
  for (int r = ty; r < 32; r += 8)
    t[r][tx] = (__bf16)in[(size_t)(by + r) * Nd + bx + tx];
  __syncthreads();
#pragma unroll
  for (int r = ty; r < 32; r += 8)
    out[(size_t)(bx + r) * Kd + by + tx] = t[tx][r];
}

// ------------------------- WMMA bf16 tiled GEMM ----------------------------
// C[M,N] (f32) = A[M,K](bf16 row-major) * Bt[N,K](bf16 row-major)^T [+ bias]
// Block tile 128x128, 8 waves (4 in M x 2 in N), wave tile 32x64 = 2x4 WMMAs
// of v_wmma_f32_16x16x32_bf16, K step 32.  Tiles staged via async global->LDS
// DMA (ASYNCcnt) with double buffering: tile k+1 copies overlap tile k WMMAs.
template <bool ADD_BIAS>
__global__ __launch_bounds__(256)
void sae_wmma_gemm_bf16(const __bf16* __restrict__ A,
                        const __bf16* __restrict__ Bt,
                        const float* __restrict__ bias,
                        float* __restrict__ C,
                        int M, int N, int K) {
  __shared__ __bf16 lsA[2][128][40];   // [buf][m][k], 80B rows (16B aligned)
  __shared__ __bf16 lsB[2][128][40];   // [buf][n][k]

  const int tid  = threadIdx.x;
  const int lane = tid & 31;
  const int wave = tid >> 5;
  const int waveM = wave & 3;          // 0..3 -> 32-row slices
  const int waveN = wave >> 2;         // 0..1 -> 64-col slices
  const int half = lane >> 4;          // ISA: lanes 0-15 vs 16-31
  const int lr   = lane & 15;

  const int m0 = blockIdx.y * 128;
  const int n0 = blockIdx.x * 128;

  // straight-line tile chunk coordinates (2 x b128 per thread per matrix)
  const int am0 = tid >> 2,          ak0 = (tid & 3) << 3;
  const int am1 = (tid + 256) >> 2,  ak1 = ((tid + 256) & 3) << 3;

  v8f acc[2][4] = {};

  // prologue: async-DMA tile 0 into buffer 0
  async_b128_to_lds(&lsA[0][am0][ak0], &A[(size_t)(m0 + am0) * K + ak0]);
  async_b128_to_lds(&lsA[0][am1][ak1], &A[(size_t)(m0 + am1) * K + ak1]);
  async_b128_to_lds(&lsB[0][am0][ak0], &Bt[(size_t)(n0 + am0) * K + ak0]);
  async_b128_to_lds(&lsB[0][am1][ak1], &Bt[(size_t)(n0 + am1) * K + ak1]);

  int cur = 0;
  for (int k0 = 0; k0 < K; k0 += 32, cur ^= 1) {
    wait_async0();        // this wave's DMA into lsA/lsB[cur] complete
    __syncthreads();      // all waves' tiles visible; prev compute on cur^1 done

    // issue next tile's DMA into the other buffer (overlaps WMMA below)
    const int kn = k0 + 32;
    if (kn < K) {
      async_b128_to_lds(&lsA[cur ^ 1][am0][ak0],
                        &A[(size_t)(m0 + am0) * K + kn + ak0]);
      async_b128_to_lds(&lsA[cur ^ 1][am1][ak1],
                        &A[(size_t)(m0 + am1) * K + kn + ak1]);
      async_b128_to_lds(&lsB[cur ^ 1][am0][ak0],
                        &Bt[(size_t)(n0 + am0) * K + kn + ak0]);
      async_b128_to_lds(&lsB[cur ^ 1][am1][ak1],
                        &Bt[(size_t)(n0 + am1) * K + kn + ak1]);
    }

    // ---- fragments per ISA lane layout ----
    // A (16-bit, 16x32): lane<16 holds K 0..7 & 16..23 of row M=lr;
    //                    lane>=16 holds K 8..15 & 24..31.
    // B (16-bit, 32x16): lane<16 holds K 0..7 & 16..23 of col N=lr; lane>=16 K 8..15 & 24..31.
    union Frag { v16bf v; bf16x8 h[2]; };
    Frag a[2], b[4];
#pragma unroll
    for (int mt = 0; mt < 2; ++mt) {
      int m = waveM * 32 + mt * 16 + lr;
      a[mt].h[0] = *(const bf16x8*)&lsA[cur][m][half * 8];
      a[mt].h[1] = *(const bf16x8*)&lsA[cur][m][16 + half * 8];
    }
#pragma unroll
    for (int nt = 0; nt < 4; ++nt) {
      int n = waveN * 64 + nt * 16 + lr;
      b[nt].h[0] = *(const bf16x8*)&lsB[cur][n][half * 8];
      b[nt].h[1] = *(const bf16x8*)&lsB[cur][n][16 + half * 8];
    }
#pragma unroll
    for (int mt = 0; mt < 2; ++mt)
#pragma unroll
      for (int nt = 0; nt < 4; ++nt)
        acc[mt][nt] = __builtin_amdgcn_wmma_f32_16x16x32_bf16(
            false, a[mt].v, false, b[nt].v, (short)0, acc[mt][nt],
            false, false);
  }

  // ---- store: C/D layout: VGPR v -> M = half*8 + v, N = lr ----
#pragma unroll
  for (int mt = 0; mt < 2; ++mt) {
#pragma unroll
    for (int nt = 0; nt < 4; ++nt) {
      int col = n0 + waveN * 64 + nt * 16 + lr;
      float bb = ADD_BIAS ? bias[col] : 0.0f;
#pragma unroll
      for (int v = 0; v < 8; ++v) {
        int row = m0 + waveM * 32 + mt * 16 + half * 8 + v;
        C[(size_t)row * N + col] = acc[mt][nt][v] + bb;
      }
    }
  }
}

// -------------------------- top-k (64) per row -----------------------------
__global__ __launch_bounds__(256)
void sae_topk_kernel(float* __restrict__ pre, float* __restrict__ vals,
                     int* __restrict__ idx) {
  __shared__ float srow[DSAE];   // 64 KB row cache (LDS is 320 KB/WGP)
  __shared__ float rv[256];
  __shared__ int   ri[256];
  const int tid = threadIdx.x;
  const int row = blockIdx.x;
  const float* p = pre + (size_t)row * DSAE;
  for (int j = tid; j < DSAE; j += 256) srow[j] = p[j];
  __syncthreads();
  for (int it = 0; it < KTOP; ++it) {
    float best = -3.0e38f; int bi = 0;
    for (int j = tid; j < DSAE; j += 256) {
      float v = srow[j];
      if (v > best) { best = v; bi = j; }
    }
    rv[tid] = best; ri[tid] = bi;
    __syncthreads();
    for (int s = 128; s > 0; s >>= 1) {
      if (tid < s && rv[tid + s] > rv[tid]) {
        rv[tid] = rv[tid + s]; ri[tid] = ri[tid + s];
      }
      __syncthreads();
    }
    if (tid == 0) {
      int w = ri[0];
      vals[(size_t)row * KTOP + it] = fmaxf(rv[0], 0.0f);   // relu(top-k vals)
      idx[(size_t)row * KTOP + it]  = w;
      srow[w] = -3.0e38f;
    }
    __syncthreads();
  }
}

// ------------------------- sparse scatter kernels --------------------------
__global__ void sae_scatter_zcur(const float* __restrict__ vals,
                                 const int* __restrict__ idx,
                                 float* __restrict__ zcur) {
  int r = blockIdx.x, j = threadIdx.x;          // r: pair, j: 0..63
  int row = 2 * r + 1;                          // token-1 rows are odd
  float v = vals[(size_t)row * KTOP + j];
  int   i = idx[(size_t)row * KTOP + j];
  zcur[(size_t)r * DSAE + i] = v;
}

__global__ void sae_norms_kernel(const float* __restrict__ vals,
                                 const int* __restrict__ idx,
                                 float* __restrict__ norms) {
  __shared__ float red[KTOP];
  int row = blockIdx.x, j = threadIdx.x;
  float v = vals[(size_t)row * KTOP + j];
  int   i = idx[(size_t)row * KTOP + j];
  red[j] = (i < HDIM) ? v * v : 0.0f;
  __syncthreads();
  for (int s = 32; s > 0; s >>= 1) {
    if (j < s) red[j] += red[j + s];
    __syncthreads();
  }
  if (j == 0) norms[row] = sqrtf(red[0]);
}

__global__ void sae_scatter_norm(const float* __restrict__ vals,
                                 const int* __restrict__ idx,
                                 const float* __restrict__ norms,
                                 __bf16* __restrict__ za,   // token-1 (cur)
                                 __bf16* __restrict__ zb) { // token-0 (prev)
  int row = blockIdx.x, j = threadIdx.x;
  float inv = 1.0f / fmaxf(norms[row], EPSN);
  float v = vals[(size_t)row * KTOP + j];
  int   i = idx[(size_t)row * KTOP + j];
  if (i < HDIM) {
    __bf16* dst = (row & 1) ? (za + (size_t)(row >> 1) * HDIM)
                            : (zb + (size_t)(row >> 1) * HDIM);
    dst[i] = (__bf16)(v * inv);
  }
}

// ------------- decoder: sparse gather + SE reduce + x_hat_cur --------------
__global__ __launch_bounds__(256)
void sae_decoder_kernel(const float* __restrict__ vals,
                        const int* __restrict__ idx,
                        const float* __restrict__ Wdec,
                        const float* __restrict__ bdec,
                        const float* __restrict__ xflat,
                        float* __restrict__ xhat_cur,
                        float* __restrict__ accums) {
  const int row = blockIdx.x;           // 0..4095
  const int tid = threadIdx.x;
  const int c0 = tid * 8;               // 8 columns of d_in per thread
  float accF[8], accH[8];
#pragma unroll
  for (int e = 0; e < 8; ++e) { float b = bdec[c0 + e]; accF[e] = b; accH[e] = b; }
  const float* vr = vals + (size_t)row * KTOP;
  const int*   ir = idx  + (size_t)row * KTOP;
  for (int j = 0; j < KTOP; ++j) {
    float v = vr[j]; int i = ir[j];
    const float* w = Wdec + (size_t)i * DIN + c0;   // W_dec L2-resident
    bool hi = (i < HDIM);
#pragma unroll
    for (int e = 0; e < 8; ++e) {
      float t = v * w[e];
      accF[e] += t;
      if (hi) accH[e] += t;
    }
  }
  const float* xr = xflat + (size_t)row * DIN + c0;
  float seF = 0.0f, seH = 0.0f;
#pragma unroll
  for (int e = 0; e < 8; ++e) {
    float d1 = accF[e] - xr[e]; seF += d1 * d1;
    float d2 = accH[e] - xr[e]; seH += d2 * d2;
  }
  if (row & 1) {
    float* o = xhat_cur + (size_t)(row >> 1) * DIN + c0;
#pragma unroll
    for (int e = 0; e < 8; ++e) o[e] = accF[e];
  }
  __shared__ float sF[256], sH[256];
  sF[tid] = seF; sH[tid] = seH;
  __syncthreads();
  for (int s = 128; s > 0; s >>= 1) {
    if (tid < s) { sF[tid] += sF[tid + s]; sH[tid] += sH[tid + s]; }
    __syncthreads();
  }
  if (tid == 0) { atomicAdd(&accums[0], sF[0]); atomicAdd(&accums[1], sH[0]); }
}

// ------------------------- logsumexp - diag -------------------------------
__global__ __launch_bounds__(256)
void sae_lse_kernel(const float* __restrict__ sim, float* __restrict__ accum,
                    int col_mode) {
  const int n = BPAIR;
  const int r = blockIdx.x;
  const int tid = threadIdx.x;
  __shared__ float red[256];
  float mx = -3.0e38f;
  for (int j = tid; j < n; j += 256) {
    float v = col_mode ? sim[(size_t)j * n + r] : sim[(size_t)r * n + j];
    mx = fmaxf(mx, v);
  }
  red[tid] = mx; __syncthreads();
  for (int s = 128; s > 0; s >>= 1) {
    if (tid < s) red[tid] = fmaxf(red[tid], red[tid + s]);
    __syncthreads();
  }
  mx = red[0]; __syncthreads();
  float sm = 0.0f;
  for (int j = tid; j < n; j += 256) {
    float v = col_mode ? sim[(size_t)j * n + r] : sim[(size_t)r * n + j];
    sm += __expf(v - mx);
  }
  red[tid] = sm; __syncthreads();
  for (int s = 128; s > 0; s >>= 1) {
    if (tid < s) red[tid] += red[tid + s];
    __syncthreads();
  }
  if (tid == 0) {
    float lse = mx + __logf(red[0]);
    float diag = sim[(size_t)r * n + r];
    atomicAdd(accum, lse - diag);
  }
}

__global__ void sae_finalize(const float* __restrict__ accums,
                             float* __restrict__ out) {
  float l_matr  = (accums[0] + accums[1]) / ((float)BPAIR * (float)DIN);
  float l_contr = 0.5f * (accums[2] + accums[3]) / (float)BPAIR;
  out[0] = l_matr + l_contr;
}

// ------------------------------- launcher ----------------------------------
extern "C" void kernel_launch(void* const* d_in, const int* in_sizes, int n_in,
                              void* d_out, int out_size, void* d_ws,
                              size_t ws_size, hipStream_t stream) {
  const float* x     = (const float*)d_in[0];   // (B,2,d_in) == (4096,2048) flat
  const float* W_enc = (const float*)d_in[1];   // (2048,16384)
  const float* W_dec = (const float*)d_in[2];   // (16384,2048)
  const float* b_enc = (const float*)d_in[3];
  const float* b_dec = (const float*)d_in[4];

  char* ws = (char*)d_ws;
  size_t off = 0;
  auto alloc = [&](size_t bytes) {
    size_t o = off;
    off += (bytes + 255) & ~(size_t)255;
    return o;
  };
  __bf16* x_bf  = (__bf16*)(ws + alloc((size_t)NROW * DIN * 2));
  __bf16* wet   = (__bf16*)(ws + alloc((size_t)DIN * DSAE * 2)); // W_enc^T [DSAE][DIN]
  float*  vals  = (float*) (ws + alloc((size_t)NROW * KTOP * 4));
  int*    idxs  = (int*)   (ws + alloc((size_t)NROW * KTOP * 4));
  float*  norms = (float*) (ws + alloc((size_t)NROW * 4));
  float*  accums= (float*) (ws + alloc(64));
  char*   big   = ws + alloc((size_t)NROW * DSAE * 4);  // 268 MB pre region
  float*  pre   = (float*)big;
  // pre is dead after top-k -> alias its space for densified z / sim
  __bf16* za    = (__bf16*)big;                                  // 33.5 MB
  __bf16* zb    = (__bf16*)(big + (size_t)BPAIR * HDIM * 2);     // 33.5 MB
  float*  sim   = (float*) (big + (size_t)BPAIR * HDIM * 4);     // 16.8 MB

  float* out_total = (float*)d_out;
  float* xhat_cur  = out_total + 1;
  float* zcur      = xhat_cur + (size_t)BPAIR * DIN;

  // 1) bf16 casts for WMMA path; W_enc cast+transpose to N-major
  sae_cast_f32_bf16<<<4096, 256, 0, stream>>>(x, x_bf, (size_t)NROW * DIN);
  dim3 gt(DSAE / 32, DIN / 32);
  sae_cast_transpose<<<gt, 256, 0, stream>>>(W_enc, wet, DIN, DSAE);

  // 2) encoder GEMM: pre = x @ W_enc + b_enc  (275 GFLOP, bf16 WMMA)
  dim3 g1(DSAE / 128, NROW / 128);
  sae_wmma_gemm_bf16<true><<<g1, 256, 0, stream>>>(
      x_bf, wet, b_enc, pre, NROW, DSAE, DIN);

  // 3) top-k(64) + relu per row
  sae_topk_kernel<<<NROW, 256, 0, stream>>>(pre, vals, idxs);

  // 4) dense z_cur output: zero + scatter
  sae_zero_f32<<<8192, 256, 0, stream>>>(zcur, (size_t)BPAIR * DSAE);
  sae_scatter_zcur<<<BPAIR, KTOP, 0, stream>>>(vals, idxs, zcur);

  // 5) decoder (sparse gather; W_dec fits L2) + SE accum + x_hat_cur
  sae_zero_f32<<<1, 64, 0, stream>>>(accums, 16);
  sae_decoder_kernel<<<NROW, 256, 0, stream>>>(vals, idxs, W_dec, b_dec, x,
                                               xhat_cur, accums);

  // 6) contrastive: normalize sparse h-slices, densify to bf16
  sae_norms_kernel<<<NROW, KTOP, 0, stream>>>(vals, idxs, norms);
  //    za+zb are contiguous: 2*BPAIR*HDIM bf16 == BPAIR*HDIM floats
  sae_zero_f32<<<4096, 256, 0, stream>>>((float*)za, (size_t)BPAIR * HDIM);
  sae_scatter_norm<<<NROW, KTOP, 0, stream>>>(vals, idxs, norms, za, zb);

  // 7) sim = za @ zb^T  (69 GFLOP, bf16 WMMA; zb rows are already N-major)
  dim3 g2(BPAIR / 128, BPAIR / 128);
  sae_wmma_gemm_bf16<false><<<g2, 256, 0, stream>>>(
      za, zb, nullptr, sim, BPAIR, BPAIR, HDIM);

  // 8) InfoNCE reductions + final scalar
  sae_lse_kernel<<<BPAIR, 256, 0, stream>>>(sim, &accums[2], 0);
  sae_lse_kernel<<<BPAIR, 256, 0, stream>>>(sim, &accums[3], 1);
  sae_finalize<<<1, 1, 0, stream>>>(accums, out_total);
}
